// GIN_23227183137278
// MI455X (gfx1250) — matrix operand order repaired and checked by reading, not compiled, not run on hardware
//
#include <hip/hip_runtime.h>
#include <hip/hip_bf16.h>

#define N_NODES 100000
#define N_EDGES 1600000

typedef __attribute__((ext_vector_type(16))) __bf16 v16bf;
typedef __attribute__((ext_vector_type(8)))  float  v8f;

union FragU { v16bf v; uint4 q[2]; };

// ---------------------------------------------------------------------------
// WMMA helper (CDNA5 wave32, layouts per ISA 7.12.2)
// ---------------------------------------------------------------------------
__device__ __forceinline__ v8f wmma_bf16(v16bf a, v16bf b, v8f c) {
    return __builtin_amdgcn_wmma_f32_16x16x32_bf16(
        false, a, false, b, (short)0, c, false, false);
}

// A 16x32 bf16 fragment from LDS tile (row stride must be multiple of 8 elems
// so both 8-element halves are 16B-aligned -> ds_load_b128).
// lane<16: M=lane, K {0..7} then {16..23}; lane>=16: M=lane-16, K {8..15},{24..31}.
__device__ __forceinline__ v16bf load_a_frag(const __bf16* __restrict__ p,
                                             int stride, int lane) {
    const int row   = lane & 15;
    const int khalf = (lane >> 4) * 8;
    const uint4* q  = (const uint4*)(p + row * stride + khalf);
    FragU u;
    u.q[0] = q[0];   // K = khalf .. khalf+7
    u.q[1] = q[2];   // K = khalf+16 .. khalf+23   (+32 bytes)
    return u.v;
}

// B fragment from pre-swizzled (fragment-major) weights: each lane's 16 bf16
// contiguous -> two coalesced global_load_b128, no packing VALU.
__device__ __forceinline__ v16bf load_b_frag(const __bf16* __restrict__ Wp,
                                             int KB, int ct, int kb32, int lane) {
    const uint4* q = (const uint4*)(Wp + (size_t)((ct * KB + kb32) * 32 + lane) * 16);
    FragU u;
    u.q[0] = q[0];
    u.q[1] = q[1];
    return u.v;
}

// ---------------------------------------------------------------------------
// Utility kernels
// ---------------------------------------------------------------------------
__global__ void zero_f32(float* __restrict__ p, int n) {
    int i = blockIdx.x * blockDim.x + threadIdx.x;
    if (i < n) p[i] = 0.0f;
}

// Convert fp32 row-major KxC weight into bf16 fragment-major layout:
// Wp[((ct*(K/32)+kb32)*32 + lane)*16 + e]  with col = ct*16 + (lane&15),
// k = kb32*32 + (lane>>4)*8 + (e&7) + (e>>3)*16   (matches B 32x16 WMMA layout).
__global__ void pack_b_bf16(const float* __restrict__ W, __bf16* __restrict__ Wp,
                            int K, int C) {
    int i = blockIdx.x * blockDim.x + threadIdx.x;
    if (i >= K * C) return;
    const int KB    = K >> 5;
    int e     = i & 15;
    int lane  = (i >> 4) & 31;
    int tile  = i >> 9;           // ct*KB + kb32
    int kb32  = tile % KB;
    int ct    = tile / KB;
    int col = ct * 16 + (lane & 15);
    int k   = kb32 * 32 + (lane >> 4) * 8 + (e & 7) + (e >> 3) * 16;
    Wp[i] = (__bf16)W[k * C + col];
}

// ---------------------------------------------------------------------------
// Edge aggregation: agg[dst] += x[src].  One wave per edge; lanes stride the
// feature dim. Node features fit in the 192MB L2, so these gathers + f32
// atomics (no return -> STOREcnt only) stay L2-resident.
// ---------------------------------------------------------------------------
__global__ __launch_bounds__(256)
void agg_kernel(const float* __restrict__ x,
                const long long* __restrict__ src,
                const long long* __restrict__ dst,
                float* __restrict__ agg, int F, int nEdges) {
    int wid  = (blockIdx.x * blockDim.x + threadIdx.x) >> 5;
    int lane = threadIdx.x & 31;
    if (wid >= nEdges) return;
    int s = (int)src[wid];
    int d = (int)dst[wid];
    const float* xs = x + (long long)s * F;
    float*       ad = agg + (long long)d * F;
    for (int f = lane; f < F; f += 32)
        atomicAdd(&ad[f], xs[f]);
}

// ---------------------------------------------------------------------------
// Fused GIN layer:  y = relu( relu(((1+eps)x + agg) @ Wa + ba) @ Wb + bb )
// Block = 256 threads (8 waves) x 32 nodes; hidden tile lives in LDS as bf16.
// ---------------------------------------------------------------------------
template <int K1, int C1, int C2>
__global__ __launch_bounds__(256)
void gin_mlp_kernel(const float* __restrict__ xin,
                    const float* __restrict__ agg,
                    const float* __restrict__ eps_p,
                    const __bf16* __restrict__ Wa, const float* __restrict__ ba,
                    const __bf16* __restrict__ Wb, const float* __restrict__ bb,
                    float* __restrict__ yout) {
    constexpr int NT  = 32;        // nodes per block (N % 32 == 0 -> full EXEC)
    constexpr int LDH = K1 + 8;    // 16B-aligned rows + bank stagger
    constexpr int LDM = C1 + 8;
    __shared__ __align__(16) __bf16 sh_h[NT * LDH];
    __shared__ __align__(16) __bf16 sh_mid[NT * LDM];

    const int tid  = threadIdx.x;
    const int lane = tid & 31;
    const int wave = tid >> 5;
    const int nb   = blockIdx.x * NT;
    const float scale = 1.0f + eps_p[0];

    // h = (1+eps)*x + agg  ->  bf16 LDS tile (paired b64 loads, b32 stores)
    for (int idx = tid; idx < NT * (K1 / 2); idx += 256) {
        int r  = idx / (K1 / 2);
        int k2 = idx - r * (K1 / 2);
        long long g = (long long)(nb + r) * K1 + 2 * k2;
        float2 xv = *(const float2*)(xin + g);
        float2 av = *(const float2*)(agg + g);
        union { __bf16 b[2]; unsigned int u; } pk;
        pk.b[0] = (__bf16)(scale * xv.x + av.x);
        pk.b[1] = (__bf16)(scale * xv.y + av.y);
        *(unsigned int*)(sh_h + r * LDH + 2 * k2) = pk.u;
    }
    __syncthreads();

    // GEMM1: mid = relu(h @ Wa + ba)
    constexpr int CT1 = C1 / 16;
    constexpr int KB1 = K1 / 32;
    for (int ct = wave; ct < CT1; ct += 8) {
        v8f acc0 = {}, acc1 = {};
#pragma unroll
        for (int kb32 = 0; kb32 < KB1; ++kb32) {
            v16bf bf = load_b_frag(Wa, KB1, ct, kb32, lane);
            v16bf a0 = load_a_frag(sh_h + kb32 * 32, LDH, lane);
            v16bf a1 = load_a_frag(sh_h + 16 * LDH + kb32 * 32, LDH, lane);
            acc0 = wmma_bf16(a0, bf, acc0);
            acc1 = wmma_bf16(a1, bf, acc1);
        }
        const int col  = ct * 16 + (lane & 15);
        const int rl   = (lane >> 4) * 8;
        const float bs = ba[col];
#pragma unroll
        for (int r = 0; r < 8; ++r) {
            sh_mid[(rl + r) * LDM + col]      = (__bf16)fmaxf(acc0[r] + bs, 0.0f);
            sh_mid[(16 + rl + r) * LDM + col] = (__bf16)fmaxf(acc1[r] + bs, 0.0f);
        }
    }
    __syncthreads();

    // GEMM2: y = relu(mid @ Wb + bb)  -> global fp32
    constexpr int CT2 = C2 / 16;
    constexpr int KB2 = C1 / 32;
    for (int ct = wave; ct < CT2; ct += 8) {
        v8f acc0 = {}, acc1 = {};
#pragma unroll
        for (int kb32 = 0; kb32 < KB2; ++kb32) {
            v16bf bf = load_b_frag(Wb, KB2, ct, kb32, lane);
            v16bf a0 = load_a_frag(sh_mid + kb32 * 32, LDM, lane);
            v16bf a1 = load_a_frag(sh_mid + 16 * LDM + kb32 * 32, LDM, lane);
            acc0 = wmma_bf16(a0, bf, acc0);
            acc1 = wmma_bf16(a1, bf, acc1);
        }
        const int col  = ct * 16 + (lane & 15);
        const int rl   = (lane >> 4) * 8;
        const float bs = bb[col];
#pragma unroll
        for (int r = 0; r < 8; ++r) {
            yout[(long long)(nb + rl + r) * C2 + col]      = fmaxf(acc0[r] + bs, 0.0f);
            yout[(long long)(nb + 16 + rl + r) * C2 + col] = fmaxf(acc1[r] + bs, 0.0f);
        }
    }
}

// ---------------------------------------------------------------------------
// FC head: out[n, 0..7] = emb[n, :] @ Wfc + bfc    (0.1 GFLOP, VALU is fine)
// ---------------------------------------------------------------------------
__global__ __launch_bounds__(256)
void fc_kernel(const float* __restrict__ emb, const float* __restrict__ W,
               const float* __restrict__ b, float* __restrict__ out, int nTot) {
    int i = blockIdx.x * blockDim.x + threadIdx.x;
    if (i >= nTot) return;
    int node = i >> 3, o = i & 7;
    const float* e = emb + (long long)node * 64;
    float acc = b[o];
#pragma unroll
    for (int k = 0; k < 64; ++k) acc += e[k] * W[k * 8 + o];
    out[i] = acc;
}

// ---------------------------------------------------------------------------
extern "C" void kernel_launch(void* const* d_in, const int* in_sizes, int n_in,
                              void* d_out, int out_size, void* d_ws, size_t ws_size,
                              hipStream_t stream) {
    (void)in_sizes; (void)n_in; (void)out_size; (void)ws_size;
    const int N = N_NODES, E = N_EDGES;

    const float*      x    = (const float*)d_in[0];
    const long long*  ei   = (const long long*)d_in[1];
    const long long*  src  = ei;
    const long long*  dst  = ei + E;
    const float* eps1 = (const float*)d_in[2];
    const float* eps2 = (const float*)d_in[3];
    const float* eps3 = (const float*)d_in[4];
    const float* W1a = (const float*)d_in[5];  const float* b1a = (const float*)d_in[6];
    const float* W1b = (const float*)d_in[7];  const float* b1b = (const float*)d_in[8];
    const float* W2a = (const float*)d_in[9];  const float* b2a = (const float*)d_in[10];
    const float* W2b = (const float*)d_in[11]; const float* b2b = (const float*)d_in[12];
    const float* W3a = (const float*)d_in[13]; const float* b3a = (const float*)d_in[14];
    const float* W3b = (const float*)d_in[15]; const float* b3b = (const float*)d_in[16];
    const float* Wfc = (const float*)d_in[17]; const float* bfc = (const float*)d_in[18];

    float* out = (float*)d_out;                 // N x 8
    float* emb = (float*)d_out + (size_t)N * 8; // N x 64  (layer-3 output)

    // Workspace layout
    float* agg = (float*)d_ws;                       // N x 128 fp32
    float* y1  = agg + (size_t)N * 128;              // N x 128 fp32
    float* y2  = y1  + (size_t)N * 128;              // N x 128 fp32
    __bf16* W1a_b = (__bf16*)(y2 + (size_t)N * 128); // bf16 fragment-major weights
    __bf16* W1b_b = W1a_b + 64 * 128;
    __bf16* W2a_b = W1b_b + 128 * 128;
    __bf16* W2b_b = W2a_b + 128 * 128;
    __bf16* W3a_b = W2b_b + 128 * 128;
    __bf16* W3b_b = W3a_b + 128 * 64;

    // Pack weights into WMMA B-fragment-major bf16 (tiny, deterministic)
    pack_b_bf16<<<(64*128  + 255) / 256, 256, 0, stream>>>(W1a, W1a_b, 64, 128);
    pack_b_bf16<<<(128*128 + 255) / 256, 256, 0, stream>>>(W1b, W1b_b, 128, 128);
    pack_b_bf16<<<(128*128 + 255) / 256, 256, 0, stream>>>(W2a, W2a_b, 128, 128);
    pack_b_bf16<<<(128*128 + 255) / 256, 256, 0, stream>>>(W2b, W2b_b, 128, 128);
    pack_b_bf16<<<(128*64  + 255) / 256, 256, 0, stream>>>(W3a, W3a_b, 128, 64);
    pack_b_bf16<<<(64*64   + 255) / 256, 256, 0, stream>>>(W3b, W3b_b, 64, 64);

    const int edgeBlocks = E / 8;   // one wave32 per edge, 8 waves/block
    const int mlpBlocks  = N / 32;  // 32 nodes per block, exact

    // ---- Layer 1: 64 -> 128 -> 128
    zero_f32<<<(N*64 + 255) / 256, 256, 0, stream>>>(agg, N*64);
    agg_kernel<<<edgeBlocks, 256, 0, stream>>>(x, src, dst, agg, 64, E);
    gin_mlp_kernel<64,128,128><<<mlpBlocks, 256, 0, stream>>>(
        x, agg, eps1, W1a_b, b1a, W1b_b, b1b, y1);

    // ---- Layer 2: 128 -> 128 -> 128
    zero_f32<<<(N*128 + 255) / 256, 256, 0, stream>>>(agg, N*128);
    agg_kernel<<<edgeBlocks, 256, 0, stream>>>(y1, src, dst, agg, 128, E);
    gin_mlp_kernel<128,128,128><<<mlpBlocks, 256, 0, stream>>>(
        y1, agg, eps2, W2a_b, b2a, W2b_b, b2b, y2);

    // ---- Layer 3: 128 -> 64 -> 64   (writes emb region of d_out directly)
    zero_f32<<<(N*128 + 255) / 256, 256, 0, stream>>>(agg, N*128);
    agg_kernel<<<edgeBlocks, 256, 0, stream>>>(y2, src, dst, agg, 128, E);
    gin_mlp_kernel<128,64,64><<<mlpBlocks, 256, 0, stream>>>(
        y2, agg, eps3, W3a_b, b3a, W3b_b, b3b, emb);

    // ---- FC head
    fc_kernel<<<(N*8 + 255) / 256, 256, 0, stream>>>(emb, Wfc, bfc, out, N*8);
}
